// Prototype_73409581023470
// MI455X (gfx1250) — matrix operand order repaired and checked by reading, not compiled, not run on hardware
//
#include <hip/hip_runtime.h>
#include <hip/hip_bf16.h>
#include <stdint.h>

// ---------------------------------------------------------------------------
// MI455X (gfx1250): NetVLAD-style multi-level aggregation. All heavy math is
// f16-in / f32-accumulate WMMA with K-contiguous fragments (global_load_b128).
// ---------------------------------------------------------------------------

typedef _Float16 half_t;
typedef __attribute__((ext_vector_type(16))) _Float16     v16h;
typedef __attribute__((ext_vector_type(8)))  float        v8f;
typedef __attribute__((ext_vector_type(4)))  unsigned int uint32x4;
typedef __attribute__((ext_vector_type(8)))  int          int32x8;
typedef __attribute__((ext_vector_type(4)))  int          int32x4;

#if defined(__gfx1250__) && __has_builtin(__builtin_amdgcn_tensor_load_to_lds)
#define HAVE_TDM 1
#else
#define HAVE_TDM 0
#endif
#if __has_include(<hip/amd_detail/amd_gfx1250_TDM.h>)
#define TDM_SIX_ARG 1
#else
#define TDM_SIX_ARG 0
#endif

#define WMMA_F16(a, b, c) \
    __builtin_amdgcn_wmma_f32_16x16x32_f16(false, (a), false, (b), (short)0, (c), false, false)

// ---- WMMA fragment index helpers (A/B use contiguous K order, not the ISA table order) ------------
// A (16x32 f16): lane&15 = M row; half i -> K = i + (lane>=16)*16  (contiguous)
__device__ __forceinline__ int a_frag_k(int lane, int i) {
    return i + ((lane >> 4) << 4);
}
// B (32x16 f16): lane&15 = N col; half i -> K = i + (lane>=16)*16  (contiguous)
// C/D (16x16 f32): vgpr v -> M = v + (lane>=16)*8 ; N = lane&15
__device__ __forceinline__ int d_frag_m(int lane, int v) {
    return v + ((lane >> 4) << 3);
}

// ---------------------------------------------------------------------------
// 1) per-pixel L2 norm. xpm [N,L,C] raw f16; xn [N,C,L] norm f16; xnT [N,L,C] norm f16
__global__ void pack_norm_kernel(const float* __restrict__ x,
                                 half_t* __restrict__ xpm,
                                 half_t* __restrict__ xn,
                                 half_t* __restrict__ xnT, int C, int L) {
    int idx = blockIdx.x * blockDim.x + threadIdx.x;
    if (idx >= 8 * L) return;
    int n = idx / L, l = idx % L;
    const float* xp = x + (size_t)n * C * L + l;
    half_t* xnp = xn + (size_t)n * C * L + l;
    half_t* xpmp = xpm + ((size_t)n * L + l) * C;
    half_t* xnTp = xnT + ((size_t)n * L + l) * C;
    float s = 0.f;
    for (int c = 0; c < C; ++c) {
        float v = xp[(size_t)c * L];
        s += v * v;
        xpmp[c] = (half_t)v;
    }
    float rinv = 1.f / fmaxf(sqrtf(s), 1e-12f);
    for (int c = 0; c < C; ++c) {
        float v = xp[(size_t)c * L] * rinv;
        xnp[(size_t)c * L] = (half_t)v;
        xnTp[c] = (half_t)v;
    }
}

// 2a) conv weights [24,C,3,3] -> per-tap A matrices [9][32,C] f16 (rows 24..31=0)
__global__ void pack_wcA_kernel(const float* __restrict__ wc,
                                half_t* __restrict__ wcA, int C) {
    int idx = blockIdx.x * blockDim.x + threadIdx.x;
    if (idx >= 9 * 32 * C) return;
    int c = idx % C;
    int r = (idx / C) % 32;
    int tap = idx / (32 * C);
    half_t v = (half_t)0.f;
    if (r < 24) v = (half_t)wc[((size_t)r * C + c) * 9 + tap];
    wcA[((size_t)tap * 32 + r) * C + c] = v;
}

// 2b) generic f32 -> f16 repack
__global__ void pack_mat_f16_kernel(const float* __restrict__ src,
                                    half_t* __restrict__ dst,
                                    int rows, int cols, int srcLd, int srcOff) {
    int idx = blockIdx.x * blockDim.x + threadIdx.x;
    if (idx >= rows * cols) return;
    int r = idx / cols, c = idx % cols;
    dst[idx] = (half_t)src[(size_t)r * srcLd + srcOff + c];
}

// ---------------------------------------------------------------------------
// 3) soft-assignment logits: implicit-GEMM 3x3 conv C->24 via 9 shifted taps.
// 64-pixel block tile; each wave: 1 weight fragment x 2 pixel fragments.
__global__ void __launch_bounds__(128)
assign_conv_kernel(const half_t* __restrict__ xnT, const half_t* __restrict__ wcA,
                   const float* __restrict__ bc, float* __restrict__ logits,
                   int C, int H, int W) {
    int L = H * W;
    int n = blockIdx.y;
    int colBase = blockIdx.x * 64;
    int wave = threadIdx.x >> 5, lane = threadIdx.x & 31;
    int wrow = wave >> 1, wcol = wave & 1;
    int m0 = wrow * 16;
    int am = m0 + (lane & 15);
    int klo = (lane >> 4) << 4;
    const half_t* xnImg = xnT + (size_t)n * L * C;
    int col0 = colBase + wcol * 16 + (lane & 15);
    int col1 = col0 + 32;
    v8f acc0 = {}, acc1 = {};
    for (int tap = 0; tap < 9; ++tap) {
        int dy = tap / 3 - 1, dx = tap % 3 - 1;
        int py0 = col0 / W + dy, px0 = col0 % W + dx;
        int py1 = col1 / W + dy, px1 = col1 % W + dx;
        bool v0 = (py0 >= 0 && py0 < H && px0 >= 0 && px0 < W);
        bool v1 = (py1 >= 0 && py1 < H && px1 >= 0 && px1 < W);
        int s0 = v0 ? (py0 * W + px0) : 0;                // clamped, always legal
        int s1 = v1 ? (py1 * W + px1) : 0;
        half_t m0h = v0 ? (half_t)1.f : (half_t)0.f;      // branchless zeroing
        half_t m1h = v1 ? (half_t)1.f : (half_t)0.f;
        const half_t* wA = wcA + (size_t)tap * 32 * C;
        for (int k0 = 0; k0 < C; k0 += 32) {
            v16h a;
#pragma unroll
            for (int i = 0; i < 16; ++i)
                a[i] = wA[(size_t)am * C + k0 + a_frag_k(lane, i)];
            v16h b0 = *(const v16h*)(xnImg + (size_t)s0 * C + k0 + klo);
            v16h b1 = *(const v16h*)(xnImg + (size_t)s1 * C + k0 + klo);
            b0 = b0 * m0h;
            b1 = b1 * m1h;
            acc0 = WMMA_F16(a, b0, acc0);
            acc1 = WMMA_F16(a, b1, acc1);
        }
    }
#pragma unroll
    for (int v = 0; v < 8; ++v) {
        int m = m0 + d_frag_m(lane, v);
        if (m < 24) {
            float bb = bc[m];
            logits[((size_t)n * 24 + m) * L + col0] = acc0[v] + bb;
            logits[((size_t)n * 24 + m) * L + col1] = acc1[v] + bb;
        }
    }
}

// 4) softmax over 24 clusters per pixel -> sa f16 [N,32,L] (rows 24..31 zero)
__global__ void softmax_kernel(const float* __restrict__ logits,
                               half_t* __restrict__ sa, int L) {
    int idx = blockIdx.x * blockDim.x + threadIdx.x;
    if (idx >= 8 * L) return;
    int n = idx / L, l = idx % L;
    const float* lg = logits + (size_t)n * 24 * L + l;
    float vals[24];
    float mx = -1e30f;
    for (int k = 0; k < 24; ++k) { vals[k] = lg[(size_t)k * L]; mx = fmaxf(mx, vals[k]); }
    float s = 0.f;
    for (int k = 0; k < 24; ++k) { vals[k] = __expf(vals[k] - mx); s += vals[k]; }
    float rs = 1.f / s;
    half_t* out = sa + (size_t)n * 32 * L + l;
    for (int k = 0; k < 24; ++k) out[(size_t)k * L] = (half_t)(vals[k] * rs);
    for (int k = 24; k < 32; ++k) out[(size_t)k * L] = (half_t)0.f;
}

// 5) ssum[n,k] = sum_l sa[n,k,l] — one wave per row, shuffle reduction
__global__ void sa_rowsum_kernel(const half_t* __restrict__ sa,
                                 float* __restrict__ ssum, int L) {
    int row = blockIdx.x * (blockDim.x >> 5) + (threadIdx.x >> 5);
    int lane = threadIdx.x & 31;
    if (row >= 8 * 32) return;
    const half_t* p = sa + (size_t)row * L;
    float s = 0.f;
    for (int l = lane; l < L; l += 32) s += (float)p[l];
#pragma unroll
    for (int off = 16; off > 0; off >>= 1) s += __shfl_down(s, off, 32);
    if (lane == 0) ssum[row] = s;
}

// 6) aggregation GEMM: up0[n,k,c] = sum_l sa[n,k,l] * xn[n,c,l]
__global__ void __launch_bounds__(128)
agg_gemm_kernel(const half_t* __restrict__ sa, const half_t* __restrict__ xn,
                float* __restrict__ up0, int C, int L) {
    int n = blockIdx.y;
    int colBase = blockIdx.x * 32;
    int wave = threadIdx.x >> 5, lane = threadIdx.x & 31;
    int wrow = wave >> 1, wcol = wave & 1;
    int m0 = wrow * 16;
    const half_t* A = sa + (size_t)n * 32 * L;
    const half_t* xi = xn + (size_t)n * C * L;
    int am = m0 + (lane & 15);
    int bn = colBase + wcol * 16 + (lane & 15);
    int klo = (lane >> 4) << 4;
    v8f acc = {};
    for (int k0 = 0; k0 < L; k0 += 32) {
        if (k0 + 32 < L) __builtin_prefetch(&A[(size_t)am * L + k0 + 32], 0, 1);
        v16h a;
#pragma unroll
        for (int i = 0; i < 16; ++i)
            a[i] = A[(size_t)am * L + k0 + a_frag_k(lane, i)];
        v16h b = *(const v16h*)(xi + (size_t)bn * L + k0 + klo);
        acc = WMMA_F16(a, b, acc);
    }
#pragma unroll
    for (int v = 0; v < 8; ++v) {
        int m = m0 + d_frag_m(lane, v);
        up0[((size_t)n * 32 + m) * C + bn] = acc[v];
    }
}

// 7) subtract centroid*ssum, intra-normalize each (n,k) row over c
__global__ void intra_norm_kernel(const float* __restrict__ up0,
                                  const float* __restrict__ cent,
                                  const float* __restrict__ ssum,
                                  float* __restrict__ upn, int C) {
    int idx = blockIdx.x * blockDim.x + threadIdx.x;
    if (idx >= 8 * 24) return;
    int n = idx / 24, k = idx % 24;
    const float* u = up0 + ((size_t)n * 32 + k) * C;
    float sm = ssum[n * 32 + k];
    float s = 0.f;
    for (int c = 0; c < C; ++c) {
        float v = u[c] - cent[(size_t)k * C + c] * sm;
        s += v * v;
    }
    float rinv = 1.f / fmaxf(sqrtf(s), 1e-12f);
    float* o = upn + ((size_t)n * 24 + k) * C;
    for (int c = 0; c < C; ++c) {
        float v = u[c] - cent[(size_t)k * C + c] * sm;
        o[c] = v * rinv;
    }
}

// 8) global L2 scale per image
__global__ void global_norm_kernel(const float* __restrict__ upn,
                                   float* __restrict__ gscale, int C) {
    __shared__ float red[256];
    int n = blockIdx.x;
    const float* u = upn + (size_t)n * 24 * C;
    int D = 24 * C;
    float s = 0.f;
    for (int i = threadIdx.x; i < D; i += 256) { float v = u[i]; s += v * v; }
    red[threadIdx.x] = s;
    __syncthreads();
    for (int o = 128; o > 0; o >>= 1) {
        if (threadIdx.x < o) red[threadIdx.x] += red[threadIdx.x + o];
        __syncthreads();
    }
    if (threadIdx.x == 0) gscale[n] = 1.f / fmaxf(sqrtf(red[0]), 1e-12f);
}

// 9a) pack upn -> f16 [16, 24C] with global-norm scale folded (rows 8..15 zero)
__global__ void pack_upn_kernel(const float* __restrict__ upn,
                                const float* __restrict__ gscale,
                                half_t* __restrict__ upnh, int D) {
    int idx = blockIdx.x * blockDim.x + threadIdx.x;
    if (idx >= 16 * D) return;
    int r = idx / D, j = idx % D;
    float v = (r < 8) ? upn[(size_t)r * D + j] * gscale[r] : 0.f;
    upnh[idx] = (half_t)v;
}

// 9b) VLAD FC as WMMA: upvec[n,c] = <upnh[n,:], wfch[c,:]> + bfc[c]
//     M=16 (8 valid images), N=64 per block, K=24C.
__global__ void __launch_bounds__(128)
fc_gemm_kernel(const half_t* __restrict__ upnh, const half_t* __restrict__ wfch,
               const float* __restrict__ bfc, float* __restrict__ upvec, int C) {
    int D = 24 * C;
    int wave = threadIdx.x >> 5, lane = threadIdx.x & 31;
    int c = blockIdx.x * 64 + wave * 16 + (lane & 15);
    int am = lane & 15;
    int klo = (lane >> 4) << 4;
    v8f acc = {};
    for (int k0 = 0; k0 < D; k0 += 32) {
        v16h a;
#pragma unroll
        for (int i = 0; i < 16; ++i)
            a[i] = upnh[(size_t)am * D + k0 + a_frag_k(lane, i)];
        v16h b = *(const v16h*)(wfch + (size_t)c * D + k0 + klo);
        acc = WMMA_F16(a, b, acc);
    }
#pragma unroll
    for (int v = 0; v < 8; ++v) {
        int n = d_frag_m(lane, v);
        if (n < 8) upvec[(size_t)n * C + c] = acc[v] + bfc[c];
    }
}

// 10) fold broadcast-up half of conv1 into per-image bias
__global__ void up_bias_kernel(const float* __restrict__ upvec,
                               const float* __restrict__ tw1,
                               const float* __restrict__ tb1,
                               float* __restrict__ bias1, int C) {
    int idx = blockIdx.x * blockDim.x + threadIdx.x;
    if (idx >= 8 * C) return;
    int n = idx / C, c = idx % C;
    const float* w = tw1 + (size_t)c * 2 * C + C;
    const float* u = upvec + (size_t)n * C;
    float acc = tb1[c];
    for (int j = 0; j < C; ++j) acc += w[j] * u[j];
    bias1[idx] = acc;
}

// ---------------------------------------------------------------------------
// 11) transform1: h[n,pix,cout] = relu( sum_k xpm[n,pix,k]*W1x[cout,k] + bias1 )
// 64x64 block; 2x2 wave grid, each wave a 32x32 quadrant (2A x 2B -> 4 WMMA).
__global__ void __launch_bounds__(128)
transform1_kernel(const half_t* __restrict__ xpm, const half_t* __restrict__ w1x,
                  const float* __restrict__ bias1, half_t* __restrict__ hpm,
                  int C, int L) {
    int n = blockIdx.z;
    int wave = threadIdx.x >> 5, lane = threadIdx.x & 31;
    int wm = wave >> 1, wn = wave & 1;
    int pixBase = blockIdx.x * 64 + wm * 32;
    int coutBase = blockIdx.y * 64 + wn * 32;
    const half_t* Ab = xpm + (size_t)n * L * C;
    int ap0 = pixBase + (lane & 15);
    int klo = (lane >> 4) << 4;
    int bcol = lane & 15;
    v8f acc00 = {}, acc01 = {}, acc10 = {}, acc11 = {};
    for (int k0 = 0; k0 < C; k0 += 32) {
        v16h a0, a1;
#pragma unroll
        for (int i = 0; i < 16; ++i) {
            int kk = k0 + a_frag_k(lane, i);
            a0[i] = Ab[(size_t)ap0 * C + kk];
            a1[i] = Ab[(size_t)(ap0 + 16) * C + kk];
        }
        v16h b0 = *(const v16h*)(w1x + (size_t)(coutBase + 0 + bcol) * C + k0 + klo);
        v16h b1 = *(const v16h*)(w1x + (size_t)(coutBase + 16 + bcol) * C + k0 + klo);
        acc00 = WMMA_F16(a0, b0, acc00);
        acc01 = WMMA_F16(a0, b1, acc01);
        acc10 = WMMA_F16(a1, b0, acc10);
        acc11 = WMMA_F16(a1, b1, acc11);
    }
    half_t* hb = hpm + (size_t)n * L * C;
#pragma unroll
    for (int ai = 0; ai < 2; ++ai) {
#pragma unroll
        for (int bi = 0; bi < 2; ++bi) {
            v8f acc = (ai == 0) ? ((bi == 0) ? acc00 : acc01)
                                : ((bi == 0) ? acc10 : acc11);
            int cout = coutBase + bi * 16 + bcol;
            float bval = bias1[(size_t)n * C + cout];
#pragma unroll
            for (int v = 0; v < 8; ++v) {
                int pix = pixBase + ai * 16 + d_frag_m(lane, v);
                float y = fmaxf(acc[v] + bval, 0.f);
                hb[(size_t)pix * C + cout] = (half_t)y;   // 32B coalesced rows
            }
        }
    }
}

// 12) transform2: out[n,cout,pix] = relu( sum_k W2[cout,k]*h[n,pix,k] + tb2 )
// Same 2x2 wave tiling; D writes NCHW directly (64B coalesced f32 rows).
// Optional TDM staging of the 64x32 B tile into LDS.
__global__ void __launch_bounds__(128)
transform2_kernel(const half_t* __restrict__ w2, const half_t* __restrict__ hpm,
                  const float* __restrict__ tb2, float* __restrict__ out,
                  int C, int L) {
    int n = blockIdx.z;
    int wave = threadIdx.x >> 5, lane = threadIdx.x & 31;
    int wm = wave >> 1, wn = wave & 1;
    int pixBlk = blockIdx.x * 64;
    int pixBase = pixBlk + wn * 32;
    int coutBase = blockIdx.y * 64 + wm * 32;
    const half_t* hb = hpm + (size_t)n * L * C;
    int am0 = coutBase + (lane & 15);
    int klo = (lane >> 4) << 4;
    int bcol = lane & 15;
    v8f acc00 = {}, acc01 = {}, acc10 = {}, acc11 = {};
#if HAVE_TDM
    __shared__ alignas(16) half_t Bs[64 * 32];   // [pixel-local][k]
#endif
    for (int k0 = 0; k0 < C; k0 += 32) {
        v16h a0, a1;
#pragma unroll
        for (int i = 0; i < 16; ++i) {
            int kk = k0 + a_frag_k(lane, i);
            a0[i] = w2[(size_t)am0 * C + kk];
            a1[i] = w2[(size_t)(am0 + 16) * C + kk];
        }
#if HAVE_TDM
        if (wave == 0) {
            const half_t* gB = hb + (size_t)pixBlk * C + k0;
            unsigned long long ga = (unsigned long long)(uintptr_t)gB;
            unsigned int la = (unsigned int)(uintptr_t)(void*)&Bs[0];
            uint32x4 g0 = { 1u, la, (unsigned int)ga,
                            (unsigned int)(ga >> 32) | (2u << 30) };
            int32x8 g1 = { (int)0x00010000u,
                           (int)((C & 0xFFFF) << 16),
                           (int)(((C >> 16) & 0xFFFF) | ((L & 0xFFFF) << 16)),
                           (int)(((L >> 16) & 0xFFFF) | (32u << 16)),
                           64, (int)C, 0, 0 };
            int32x4 g2 = { 0, 0, 0, 0 };
            int32x4 g3 = { 0, 0, 0, 0 };
#if TDM_SIX_ARG
            int32x8 g4 = { 0, 0, 0, 0, 0, 0, 0, 0 };
            __builtin_amdgcn_tensor_load_to_lds(g0, g1, g2, g3, g4, 0);
#else
            __builtin_amdgcn_tensor_load_to_lds(g0, g1, g2, g3, 0);
#endif
#if __has_builtin(__builtin_amdgcn_s_wait_tensorcnt)
            __builtin_amdgcn_s_wait_tensorcnt(0);
#endif
        }
        __syncthreads();
        v16h b0 = *(const v16h*)&Bs[(size_t)(wn * 32 + 0 + bcol) * 32 + klo];
        v16h b1 = *(const v16h*)&Bs[(size_t)(wn * 32 + 16 + bcol) * 32 + klo];
#else
        if (k0 + 32 < C)
            __builtin_prefetch(hb + (size_t)(pixBase + bcol) * C + k0 + 32, 0, 1);
        v16h b0 = *(const v16h*)(hb + (size_t)(pixBase + 0 + bcol) * C + k0 + klo);
        v16h b1 = *(const v16h*)(hb + (size_t)(pixBase + 16 + bcol) * C + k0 + klo);
#endif
        acc00 = WMMA_F16(a0, b0, acc00);
        acc01 = WMMA_F16(a0, b1, acc01);
        acc10 = WMMA_F16(a1, b0, acc10);
        acc11 = WMMA_F16(a1, b1, acc11);
#if HAVE_TDM
        __syncthreads();
#endif
    }
#pragma unroll
    for (int ai = 0; ai < 2; ++ai) {
#pragma unroll
        for (int bi = 0; bi < 2; ++bi) {
            v8f acc = (ai == 0) ? ((bi == 0) ? acc00 : acc01)
                                : ((bi == 0) ? acc10 : acc11);
            int pix = pixBase + bi * 16 + bcol;
#pragma unroll
            for (int v = 0; v < 8; ++v) {
                int cout = coutBase + ai * 16 + d_frag_m(lane, v);
                float y = fmaxf(acc[v] + tb2[cout], 0.f);
                out[((size_t)n * C + cout) * L + pix] = y;   // 64B coalesced
            }
        }
    }
}

// ---------------------------------------------------------------------------
extern "C" void kernel_launch(void* const* d_in, const int* in_sizes, int n_in,
                              void* d_out, int out_size, void* d_ws, size_t ws_size,
                              hipStream_t stream) {
    (void)in_sizes; (void)n_in; (void)out_size; (void)ws_size;
    const int N = 8;
    struct Lvl { int C, H, W, fIdx; size_t outOff; };
    const Lvl lv[4] = {
        {512, 16, 16, 3, 15204352u},
        {320, 32, 32, 2, 12582912u},
        {128, 64, 64, 1, 8388608u},
        {64, 128, 128, 0, 0u},
    };
    char* ws = (char*)d_ws;
    size_t off = 0;
    auto alloc = [&](size_t b) { size_t r = off; off += (b + 255) & ~(size_t)255; return r; };
    const size_t MAX_NCL = (size_t)N * 64 * 16384;
    const size_t MAX_L = 16384;
    const size_t MAX_C = 512;
    const size_t MAX_D = 24 * MAX_C;
    half_t* xpm   = (half_t*)(ws + alloc(MAX_NCL * 2));
    half_t* xn    = (half_t*)(ws + alloc(MAX_NCL * 2));
    half_t* xnT   = (half_t*)(ws + alloc(MAX_NCL * 2));
    half_t* hpm   = (half_t*)(ws + alloc(MAX_NCL * 2));
    half_t* sa    = (half_t*)(ws + alloc((size_t)N * 32 * MAX_L * 2));
    float*  logit = (float*)(ws + alloc((size_t)N * 24 * MAX_L * 4));
    half_t* wcA   = (half_t*)(ws + alloc((size_t)9 * 32 * MAX_C * 2));
    half_t* w1x   = (half_t*)(ws + alloc(MAX_C * MAX_C * 2));
    half_t* w2f   = (half_t*)(ws + alloc(MAX_C * MAX_C * 2));
    half_t* wfch  = (half_t*)(ws + alloc(MAX_C * MAX_D * 2));
    half_t* upnh  = (half_t*)(ws + alloc((size_t)16 * MAX_D * 2));
    float*  up0   = (float*)(ws + alloc((size_t)N * 32 * MAX_C * 4));
    float*  ssum  = (float*)(ws + alloc((size_t)N * 32 * 4));
    float*  upn   = (float*)(ws + alloc((size_t)N * 24 * MAX_C * 4));
    float*  gscale= (float*)(ws + alloc((size_t)N * 4));
    float*  upvec = (float*)(ws + alloc((size_t)N * MAX_C * 4));
    float*  bias1 = (float*)(ws + alloc((size_t)N * MAX_C * 4));

    const int tpb = 256;
    for (int li = 0; li < 4; ++li) {
        int C = lv[li].C, H = lv[li].H, Wd = lv[li].W, L = H * Wd;
        int D = 24 * C;
        const float* x = (const float*)d_in[lv[li].fIdx];
        int pb = 4 + li * 9;
        const float* wc   = (const float*)d_in[pb + 0];
        const float* bc   = (const float*)d_in[pb + 1];
        const float* cent = (const float*)d_in[pb + 2];
        const float* wfc  = (const float*)d_in[pb + 3];
        const float* bfc  = (const float*)d_in[pb + 4];
        const float* tw1  = (const float*)d_in[pb + 5];
        const float* tb1  = (const float*)d_in[pb + 6];
        const float* tw2  = (const float*)d_in[pb + 7];
        const float* tb2  = (const float*)d_in[pb + 8];
        float* out = (float*)d_out + lv[li].outOff;

        pack_norm_kernel<<<dim3((N * L + tpb - 1) / tpb), tpb, 0, stream>>>(x, xpm, xn, xnT, C, L);
        pack_wcA_kernel<<<dim3((9 * 32 * C + tpb - 1) / tpb), tpb, 0, stream>>>(wc, wcA, C);
        pack_mat_f16_kernel<<<dim3((C * C + tpb - 1) / tpb), tpb, 0, stream>>>(tw1, w1x, C, C, 2 * C, 0);
        pack_mat_f16_kernel<<<dim3((C * C + tpb - 1) / tpb), tpb, 0, stream>>>(tw2, w2f, C, C, C, 0);
        pack_mat_f16_kernel<<<dim3((C * D + tpb - 1) / tpb), tpb, 0, stream>>>(wfc, wfch, C, D, D, 0);
        assign_conv_kernel<<<dim3(L / 64, N), 128, 0, stream>>>(xnT, wcA, bc, logit, C, H, Wd);
        softmax_kernel<<<dim3((N * L + tpb - 1) / tpb), tpb, 0, stream>>>(logit, sa, L);
        sa_rowsum_kernel<<<dim3(32), 256, 0, stream>>>(sa, ssum, L);
        agg_gemm_kernel<<<dim3(C / 32, N), 128, 0, stream>>>(sa, xn, up0, C, L);
        intra_norm_kernel<<<dim3(1), 256, 0, stream>>>(up0, cent, ssum, upn, C);
        global_norm_kernel<<<dim3(N), 256, 0, stream>>>(upn, gscale, C);
        pack_upn_kernel<<<dim3((16 * D + tpb - 1) / tpb), tpb, 0, stream>>>(upn, gscale, upnh, D);
        fc_gemm_kernel<<<dim3(C / 64), 128, 0, stream>>>(upnh, wfch, bfc, upvec, C);
        up_bias_kernel<<<dim3((N * C + tpb - 1) / tpb), tpb, 0, stream>>>(upvec, tw1, tb1, bias1, C);
        transform1_kernel<<<dim3(L / 64, C / 64, N), 128, 0, stream>>>(xpm, w1x, bias1, hpm, C, L);
        transform2_kernel<<<dim3(L / 64, C / 64, N), 128, 0, stream>>>(w2f, hpm, tb2, out, C, L);
    }
}
